// KascadeReuseAttention_53386443489643
// MI455X (gfx1250) — matrix operand order repaired and checked by reading, not compiled, not run on hardware
//
#include <hip/hip_runtime.h>
#include <hip/hip_bf16.h>

// ---------------------------------------------------------------------------
// KascadeReuseAttention for MI455X (gfx1250, wave32, WMMA)
// [qkv+rope GEMM] -> [sparse flash-attention over gathered tiles] -> [o-proj]
// f16 operands / f32 accumulation via v_wmma_f32_16x16x32_f16.
// All LDS B-operands stored [n][k] so every fragment read is 2x ds_load_b128.
// Staging loops use explicit iteration counts and split gather/convert phases
// so global loads issue back-to-back under s_clause.
// ---------------------------------------------------------------------------

typedef _Float16 half_t;
typedef __attribute__((ext_vector_type(16))) _Float16 v16h;
typedef __attribute__((ext_vector_type(8)))  _Float16 v8h;
typedef __attribute__((ext_vector_type(4)))  _Float16 v4h;
typedef __attribute__((ext_vector_type(4)))  float    v4f;
typedef __attribute__((ext_vector_type(8)))  float    v8f;
typedef __attribute__((ext_vector_type(4)))  int      v4i;

#define BATCH   2
#define SEQ     4096
#define HEADS   16
#define HD      128
#define DMODEL  2048
#define NTILE   32
#define TILE_Q  128
#define TOPK    7
#define NEGBIG  (-1e10f)

// LDS strides in half elements.
#define AST 40    // A tiles: [128 m][32 k] padded
#define WST 40    // B tiles (transposed): [128 n][32 k] padded
#define YST 132   // f32 epilogue tile
#define QST 136   // [128][128] tiles, multiple of 8 for 16B alignment
#define KST 136
#define PST 136

#define K1_LDS (128*AST*2 + 128*WST*2 + 128*YST*4)
#define K2_LDS (128*QST*2 + 128*KST*2 + 8*16*PST*2)
#define K3_LDS (128*AST*2 + 128*WST*2)

// ---- CDNA5 async global->LDS copies (guarded; fallback = v8h copy) --------
#if defined(__has_builtin)
#if __has_builtin(__builtin_amdgcn_global_load_async_to_lds_b128) && \
    __has_builtin(__builtin_amdgcn_s_wait_asynccnt)
#define USE_ASYNC 1
#endif
#endif
#ifndef USE_ASYNC
#define USE_ASYNC 0
#endif

#if USE_ASYNC
typedef __attribute__((address_space(1))) v4i* gv4i_p;
typedef __attribute__((address_space(3))) v4i* lv4i_p;
#endif

static __device__ __forceinline__ void cp16(half_t* l, const half_t* g) {
#if USE_ASYNC
  __builtin_amdgcn_global_load_async_to_lds_b128(
      (gv4i_p)(half_t*)g, (lv4i_p)l, 0, 0);
#else
  *(v8h*)l = *(const v8h*)g;
#endif
}
static __device__ __forceinline__ void cp_wait() {
#if USE_ASYNC
  __builtin_amdgcn_s_wait_asynccnt(0);
#endif
}

// ---------------------------------------------------------------------------
static __device__ __forceinline__ v8f wmma_f16(v16h a, v16h b, v8f c) {
  return __builtin_amdgcn_wmma_f32_16x16x32_f16(false, a, false, b, (short)0, c,
                                                false, false);
}

// A fragment: 16x32 f16 from [m][k] tile. lane&15=M; lane>>4 picks k-halves.
// half h -> k = (lane>>4)*16 + h  -> contiguous 32B.
static __device__ __forceinline__ v16h frag_a(const half_t* base, int stride,
                                              int m0, int lane) {
  v16h a;
  const half_t* p = base + (size_t)(m0 + (lane & 15)) * stride + ((lane >> 4) << 4);
#pragma unroll
  for (int h = 0; h < 16; ++h) a[h] = p[h];
  return a;
}

// B fragment: 32x16 f16 from an [n][k] (transposed) tile: B[k][n]=src[n*st+k].
// lane&15 = N col; half h -> k = k0 + (lane>>4)*16 + h  -> contiguous 32B.
static __device__ __forceinline__ v16h frag_bt(const half_t* base, int stride,
                                               int n0, int k0, int lane) {
  v16h b;
  const half_t* p = base + (size_t)(n0 + (lane & 15)) * stride + k0 + ((lane >> 4) << 4);
#pragma unroll
  for (int h = 0; h < 16; ++h) b[h] = p[h];
  return b;
}

// ---------------------------------------------------------------------------
// Kernel 1: Y = X @ W{q,k,v}; RoPE on q,k; 1/sqrt(D) folded into q.
// grid = (HEADS, 64, 3), block = 256 (8 waves, 2x4 wave grid over 128x128).
// ---------------------------------------------------------------------------
__global__ __launch_bounds__(256)
void qkv_rope_kernel(const float* __restrict__ x,
                     const float* __restrict__ wq,
                     const float* __restrict__ wk,
                     const float* __restrict__ wv,
                     const float* __restrict__ cost,
                     const float* __restrict__ sint,
                     half_t* __restrict__ Qh,
                     half_t* __restrict__ Kh,
                     half_t* __restrict__ Vh) {
  extern __shared__ char smem[];
  half_t* a_lds = (half_t*)smem;                          // [128][AST]
  half_t* b_lds = (half_t*)(smem + 128 * AST * 2);        // [128 n][WST]
  float*  y_lds = (float*)(smem + 128 * AST * 2 + 128 * WST * 2); // [128][YST]

  const int tid  = threadIdx.x;
  const int lane = tid & 31;
  const int wid  = tid >> 5;
  const int wm   = wid >> 2;
  const int wn   = wid & 3;
  const int head = blockIdx.x;
  const int mb   = blockIdx.y;
  const int which = blockIdx.z;   // 0=q 1=k 2=v
  const float* w = (which == 0) ? wq : (which == 1) ? wk : wv;
  const int row0 = mb * 128;
  const int col0 = head * 128;

  v8f acc[4][2];
#pragma unroll
  for (int i = 0; i < 4; ++i)
#pragma unroll
    for (int j = 0; j < 2; ++j) acc[i][j] = (v8f)0.f;

  for (int ks = 0; ks < DMODEL; ks += 32) {
    __syncthreads();
    // Gather phase: 4 A-slices + 4 B-slices loaded back-to-back.
    v4f av[4], wv4[4];
#pragma unroll
    for (int it = 0; it < 4; ++it) {
      int e = tid + it * 256;
      int m = e >> 3, k4 = (e & 7) * 4;
      av[it] = *(const v4f*)&x[(size_t)(row0 + m) * DMODEL + ks + k4];
    }
#pragma unroll
    for (int it = 0; it < 4; ++it) {
      int e = tid + it * 256;
      int k = e >> 5, n4 = (e & 31) * 4;
      wv4[it] = *(const v4f*)&w[(size_t)(ks + k) * DMODEL + col0 + n4];
    }
    // Convert + store phase.
#pragma unroll
    for (int it = 0; it < 4; ++it) {
      int e = tid + it * 256;
      int m = e >> 3, k4 = (e & 7) * 4;
      v4h hv;
#pragma unroll
      for (int i = 0; i < 4; ++i) hv[i] = (half_t)av[it][i];
      *(v4h*)&a_lds[m * AST + k4] = hv;
    }
#pragma unroll
    for (int it = 0; it < 4; ++it) {
      int e = tid + it * 256;
      int k = e >> 5, n4 = (e & 31) * 4;
#pragma unroll
      for (int i = 0; i < 4; ++i) b_lds[(n4 + i) * WST + k] = (half_t)wv4[it][i];
    }
    if (ks + 32 < DMODEL) {   // prefetch next k-slice -> global_prefetch_b8
      __builtin_prefetch(&x[(size_t)(row0 + (tid >> 1)) * DMODEL + ks + 32 + (tid & 1) * 16], 0, 1);
      __builtin_prefetch(&w[(size_t)(ks + 32 + (tid >> 3)) * DMODEL + col0 + (tid & 7) * 16], 0, 1);
    }
    __syncthreads();

    v16h bf[2];
#pragma unroll
    for (int j = 0; j < 2; ++j) bf[j] = frag_bt(b_lds, WST, wn * 32 + j * 16, 0, lane);
#pragma unroll
    for (int i = 0; i < 4; ++i) {
      v16h af = frag_a(a_lds, AST, wm * 64 + i * 16, lane);
#pragma unroll
      for (int j = 0; j < 2; ++j) acc[i][j] = wmma_f16(af, bf[j], acc[i][j]);
    }
  }

  // Spill accumulators to LDS so RoPE can pair columns d and d+-64.
  const int lhalf = lane >> 4, lidx = lane & 15;
#pragma unroll
  for (int i = 0; i < 4; ++i)
#pragma unroll
    for (int j = 0; j < 2; ++j)
#pragma unroll
      for (int r = 0; r < 8; ++r) {
        int m = wm * 64 + i * 16 + r + lhalf * 8;
        int n = wn * 32 + j * 16 + lidx;
        y_lds[m * YST + n] = acc[i][j][r];
      }
  __syncthreads();

  const float qscale = 0.08838834764831845f; // 1/sqrt(128)
  half_t* dst = (which == 0) ? Qh : (which == 1) ? Kh : Vh;
  // Rolled (x4) epilogue: streaming stores, no benefit from full unroll.
#pragma unroll 4
  for (int it = 0; it < 64; ++it) {
    int e = tid + it * 256;
    int m = e >> 7, d = e & 127;
    int r = row0 + m;
    int b = r >> 12;
    int s = r & (SEQ - 1);
    float val = y_lds[m * YST + d];
    float outv;
    if (which < 2) {
      float other = y_lds[m * YST + ((d + 64) & 127)];
      float rot = (d < 64) ? -other : other;   // rotate_half
      float c  = cost[s * 64 + (d & 63)];
      float sn = sint[s * 64 + (d & 63)];
      outv = val * c + rot * sn;
      if (which == 0) outv *= qscale;
    } else {
      outv = val;
    }
    dst[(((size_t)b * HEADS + head) * SEQ + s) * HD + d] = (half_t)outv;
  }
}

// ---------------------------------------------------------------------------
// Kernel 2: sparse flash attention over 8 gathered tiles (7 anchors + local).
// grid = (NTILE, HEADS, BATCH), block = 256 (wave w owns q rows [16w,16w+16)).
// Q/K tiles copied with async global->LDS b128; V staged transposed [d][key]
// into the same buffer so the PV B-fragment read is contiguous too.
// ---------------------------------------------------------------------------
__global__ __launch_bounds__(256)
void attn_kernel(const half_t* __restrict__ Qh,
                 const half_t* __restrict__ Kh,
                 const half_t* __restrict__ Vh,
                 const int* __restrict__ anchors,
                 half_t* __restrict__ AOh) {
  extern __shared__ char smem[];
  half_t* q_lds  = (half_t*)smem;                 // [128][QST]
  half_t* kv_lds = q_lds + 128 * QST;             // K:[key][d] then VT:[d][key]
  half_t* p_lds  = kv_lds + 128 * KST;            // [8 waves][16][PST]

  const int tid  = threadIdx.x;
  const int lane = tid & 31;
  const int wid  = tid >> 5;
  const int lhalf = lane >> 4, lidx = lane & 15;
  const int qt = blockIdx.x;
  const int h  = blockIdx.y;
  const int b  = blockIdx.z;
  const size_t headoff = (((size_t)b * HEADS + h) * SEQ) * HD;

  // Async-load the Q tile (8 x b128 per thread).
#pragma unroll
  for (int it = 0; it < 8; ++it) {
    int e = tid + it * 256;
    int m = e >> 4, d8 = (e & 15) * 8;
    cp16(&q_lds[m * QST + d8], &Qh[headoff + (size_t)(qt * TILE_Q + m) * HD + d8]);
  }
  cp_wait();
  __syncthreads();

  v16h aq[4];
#pragma unroll
  for (int c = 0; c < 4; ++c) aq[c] = frag_a(q_lds + c * 32, QST, wid * 16, lane);

  v8f o[8];
#pragma unroll
  for (int j = 0; j < 8; ++j) o[j] = (v8f)0.f;
  float mrow[8], lrow[8];
#pragma unroll
  for (int r = 0; r < 8; ++r) { mrow[r] = -__builtin_inff(); lrow[r] = 0.f; }

  half_t* pw = p_lds + wid * 16 * PST;

  for (int kt = 0; kt < 8; ++kt) {
    const int tile = (kt < TOPK)
        ? anchors[(((b * HEADS + h) * NTILE) + qt) * TOPK + kt]
        : qt;

    __syncthreads();   // previous iteration done with kv_lds
    // Async-load K tile [key][d].
#pragma unroll
    for (int it = 0; it < 8; ++it) {
      int e = tid + it * 256;
      int m = e >> 4, d8 = (e & 15) * 8;
      cp16(&kv_lds[m * KST + d8],
           &Kh[headoff + (size_t)(tile * TILE_Q + m) * HD + d8]);
    }
    cp_wait();
    // Prefetch this tile's V while QK^T runs.
    __builtin_prefetch(&Vh[headoff + (size_t)(tile * TILE_Q + (tid >> 1)) * HD + (tid & 1) * 64], 0, 1);
    __syncthreads();

    // S = Q @ K^T  (8 key-subtiles x 4 k-steps = 32 WMMAs)
    v8f sf[8];
#pragma unroll
    for (int j = 0; j < 8; ++j) {
      v8f s = (v8f)0.f;
#pragma unroll
      for (int c = 0; c < 4; ++c) {
        v16h bk = frag_bt(kv_lds, KST, j * 16, c * 32, lane);
        s = wmma_f16(aq[c], bk, s);
      }
      sf[j] = s;
    }

    // Causal mask.
#pragma unroll
    for (int j = 0; j < 8; ++j)
#pragma unroll
      for (int r = 0; r < 8; ++r) {
        int qpos = qt * TILE_Q + wid * 16 + r + lhalf * 8;
        int kpos = tile * TILE_Q + j * 16 + lidx;
        if (kpos > qpos) sf[j][r] = NEGBIG;
      }

    // Online softmax.
#pragma unroll
    for (int r = 0; r < 8; ++r) {
      float mx = sf[0][r];
#pragma unroll
      for (int j = 1; j < 8; ++j) mx = fmaxf(mx, sf[j][r]);
#pragma unroll
      for (int off = 1; off < 16; off <<= 1) mx = fmaxf(mx, __shfl_xor(mx, off, 32));
      float mnew  = fmaxf(mrow[r], mx);
      float alpha = __expf(mrow[r] - mnew);
      mrow[r] = mnew;
      float rs = 0.f;
#pragma unroll
      for (int j = 0; j < 8; ++j) {
        float p = __expf(sf[j][r] - mnew);
        sf[j][r] = p;
        rs += p;
      }
#pragma unroll
      for (int off = 1; off < 16; off <<= 1) rs += __shfl_xor(rs, off, 32);
      lrow[r] = lrow[r] * alpha + rs;
#pragma unroll
      for (int j = 0; j < 8; ++j) o[j][r] *= alpha;
    }

    // P (C-layout) -> wave-private LDS -> A-layout fragments.
#pragma unroll
    for (int j = 0; j < 8; ++j)
#pragma unroll
      for (int r = 0; r < 8; ++r) {
        int m = r + lhalf * 8;
        pw[m * PST + j * 16 + lidx] = (half_t)sf[j][r];
      }
    v16h ap[4];
#pragma unroll
    for (int c = 0; c < 4; ++c) ap[c] = frag_a(pw + c * 32, PST, 0, lane);

    __syncthreads();   // all waves done reading K
    // Stage V transposed: kv_lds[d][key] = V[key][d].  Gather then scatter.
    v8h vv[8];
#pragma unroll
    for (int it = 0; it < 8; ++it) {
      int e = tid + it * 256;
      int key = e >> 4, d8 = (e & 15) * 8;
      vv[it] = *(const v8h*)&Vh[headoff + (size_t)(tile * TILE_Q + key) * HD + d8];
    }
#pragma unroll
    for (int it = 0; it < 8; ++it) {
      int e = tid + it * 256;
      int key = e >> 4, d8 = (e & 15) * 8;
#pragma unroll
      for (int i = 0; i < 8; ++i) kv_lds[(d8 + i) * KST + key] = vv[it][i];
    }
    __syncthreads();

    // O += P @ V  (B[k=key][n=d] = VT[n*KST + k], contiguous reads)
#pragma unroll
    for (int j = 0; j < 8; ++j)
#pragma unroll
      for (int c = 0; c < 4; ++c) {
        v16h bv = frag_bt(kv_lds, KST, j * 16, c * 32, lane);
        o[j] = wmma_f16(ap[c], bv, o[j]);
      }
  }

  // Normalize; store attention output f16 [B, S, H*D].
#pragma unroll
  for (int r = 0; r < 8; ++r) {
    float inv = (lrow[r] > 0.f) ? (1.f / lrow[r]) : 0.f;
#pragma unroll
    for (int j = 0; j < 8; ++j) {
      int m = wid * 16 + r + lhalf * 8;
      int s = qt * TILE_Q + m;
      int d = j * 16 + lidx;
      AOh[((size_t)b * SEQ + s) * DMODEL + h * HD + d] = (half_t)(o[j][r] * inv);
    }
  }
}

// ---------------------------------------------------------------------------
// Kernel 3: out = attn @ wo (f32 output).
// ---------------------------------------------------------------------------
__global__ __launch_bounds__(256)
void oproj_kernel(const half_t* __restrict__ AOh,
                  const float* __restrict__ wo,
                  float* __restrict__ out) {
  extern __shared__ char smem[];
  half_t* a_lds = (half_t*)smem;                    // [128][AST]
  half_t* b_lds = (half_t*)(smem + 128 * AST * 2);  // [128 n][WST]

  const int tid  = threadIdx.x;
  const int lane = tid & 31;
  const int wid  = tid >> 5;
  const int wm   = wid >> 2;
  const int wn   = wid & 3;
  const int nb   = blockIdx.x;
  const int mb   = blockIdx.y;
  const int row0 = mb * 128;
  const int col0 = nb * 128;

  v8f acc[4][2];
#pragma unroll
  for (int i = 0; i < 4; ++i)
#pragma unroll
    for (int j = 0; j < 2; ++j) acc[i][j] = (v8f)0.f;

  for (int ks = 0; ks < DMODEL; ks += 32) {
    __syncthreads();
    // Gather: A (already f16) 2 x b128 + B 4 x float4, then stores.
    v8h ah[2];
    v4f wv4[4];
#pragma unroll
    for (int it = 0; it < 2; ++it) {
      int e = tid + it * 256;
      int m = e >> 2, k8 = (e & 3) * 8;
      ah[it] = *(const v8h*)&AOh[(size_t)(row0 + m) * DMODEL + ks + k8];
    }
#pragma unroll
    for (int it = 0; it < 4; ++it) {
      int e = tid + it * 256;
      int k = e >> 5, n4 = (e & 31) * 4;
      wv4[it] = *(const v4f*)&wo[(size_t)(ks + k) * DMODEL + col0 + n4];
    }
#pragma unroll
    for (int it = 0; it < 2; ++it) {
      int e = tid + it * 256;
      int m = e >> 2, k8 = (e & 3) * 8;
      *(v8h*)&a_lds[m * AST + k8] = ah[it];
    }
#pragma unroll
    for (int it = 0; it < 4; ++it) {
      int e = tid + it * 256;
      int k = e >> 5, n4 = (e & 31) * 4;
#pragma unroll
      for (int i = 0; i < 4; ++i) b_lds[(n4 + i) * WST + k] = (half_t)wv4[it][i];
    }
    if (ks + 32 < DMODEL) {
      __builtin_prefetch(&AOh[(size_t)(row0 + (tid >> 1)) * DMODEL + ks + 32 + (tid & 1) * 16], 0, 1);
      __builtin_prefetch(&wo[(size_t)(ks + 32 + (tid >> 3)) * DMODEL + col0 + (tid & 7) * 16], 0, 1);
    }
    __syncthreads();

    v16h bf[2];
#pragma unroll
    for (int j = 0; j < 2; ++j) bf[j] = frag_bt(b_lds, WST, wn * 32 + j * 16, 0, lane);
#pragma unroll
    for (int i = 0; i < 4; ++i) {
      v16h af = frag_a(a_lds, AST, wm * 64 + i * 16, lane);
#pragma unroll
      for (int j = 0; j < 2; ++j) acc[i][j] = wmma_f16(af, bf[j], acc[i][j]);
    }
  }

  const int lhalf = lane >> 4, lidx = lane & 15;
#pragma unroll
  for (int i = 0; i < 4; ++i)
#pragma unroll
    for (int j = 0; j < 2; ++j)
#pragma unroll
      for (int r = 0; r < 8; ++r) {
        int m = wm * 64 + i * 16 + r + lhalf * 8;
        int n = wn * 32 + j * 16 + lidx;
        out[(size_t)(row0 + m) * DMODEL + col0 + n] = acc[i][j][r];
      }
}

// ---------------------------------------------------------------------------
extern "C" void kernel_launch(void* const* d_in, const int* in_sizes, int n_in,
                              void* d_out, int out_size, void* d_ws, size_t ws_size,
                              hipStream_t stream) {
  (void)in_sizes; (void)n_in; (void)out_size; (void)ws_size;
  const float* x    = (const float*)d_in[0];
  const float* wq   = (const float*)d_in[1];
  const float* wk   = (const float*)d_in[2];
  const float* wv   = (const float*)d_in[3];
  const float* wo   = (const float*)d_in[4];
  const float* cost = (const float*)d_in[5];
  const float* sint = (const float*)d_in[6];
  const int* anchors = (const int*)d_in[7];
  float* out = (float*)d_out;

  const size_t per_tensor = (size_t)BATCH * HEADS * SEQ * HD;
  half_t* Qh  = (half_t*)d_ws;
  half_t* Kh  = Qh + per_tensor;
  half_t* Vh  = Kh + per_tensor;
  half_t* AOh = Vh + per_tensor;

  dim3 blk(256);
  qkv_rope_kernel<<<dim3(HEADS, (BATCH * SEQ) / 128, 3), blk, K1_LDS, stream>>>(
      x, wq, wk, wv, cost, sint, Qh, Kh, Vh);
  attn_kernel<<<dim3(NTILE, HEADS, BATCH), blk, K2_LDS, stream>>>(
      Qh, Kh, Vh, anchors, AOh);
  oproj_kernel<<<dim3(DMODEL / 128, (BATCH * SEQ) / 128), blk, K3_LDS, stream>>>(
      AOh, wo, out);
}
